// PPNet_23862838297304
// MI455X (gfx1250) — matrix operand order repaired and checked by reading, not compiled, not run on hardware
//
#include <hip/hip_runtime.h>

typedef _Float16 v8h  __attribute__((ext_vector_type(8)));
typedef _Float16 v16h __attribute__((ext_vector_type(16)));
typedef float    v8f  __attribute__((ext_vector_type(8)));

#define B_ROWS 32768
#define D_DOM  4
#define NFEAT  8
#define E_DIM  16
#define VOCAB  100000

// ---------------------------------------------------------------------------
// CDNA5 async global->LDS copy (ASYNCcnt path, ISA 08_async_tensor §4)
// ---------------------------------------------------------------------------
__device__ __forceinline__ void async_copy_b128(unsigned lds_byte, const void* gptr)
{
    asm volatile("global_load_async_to_lds_b128 %0, %1, off"
                 :: "v"(lds_byte), "v"((unsigned long long)(uintptr_t)gptr)
                 : "memory");
}
__device__ __forceinline__ void wait_asynccnt0()
{
    asm volatile("s_wait_asynccnt 0x0" ::: "memory");
}

__device__ __forceinline__ v16h load_v16h(const _Float16* p)
{
    v8h lo = *(const v8h*)p;
    v8h hi = *(const v8h*)(p + 8);
    return __builtin_shufflevector(lo, hi,
           0, 1, 2, 3, 4, 5, 6, 7, 8, 9, 10, 11, 12, 13, 14, 15);
}

// ---------------------------------------------------------------------------
// Embedding gather -> gate_in f16 [B][256]  (cols 0..127 id, 128..255 agn)
// ---------------------------------------------------------------------------
__global__ void ppnet_embed(const int* __restrict__ id_idx,
                            const int* __restrict__ agn_idx,
                            const float* __restrict__ id_tables,
                            const float* __restrict__ agn_tables,
                            _Float16* __restrict__ gate_in)
{
    int t = blockIdx.x * blockDim.x + threadIdx.x;       // t = b*NFEAT + f
    if (t >= B_ROWS * NFEAT) return;
    int b = t / NFEAT;
    int f = t % NFEAT;

    const float* idrow = id_tables  + ((size_t)f * VOCAB + id_idx[t])  * E_DIM;
    const float* agrow = agn_tables + ((size_t)f * VOCAB + agn_idx[t]) * E_DIM;
    _Float16* o1 = gate_in + (size_t)b * 256 + f * E_DIM;
    _Float16* o2 = gate_in + (size_t)b * 256 + 128 + f * E_DIM;
#pragma unroll
    for (int e = 0; e < E_DIM; ++e) o1[e] = (_Float16)idrow[e];
#pragma unroll
    for (int e = 0; e < E_DIM; ++e) o2[e] = (_Float16)agrow[e];
}

// ---------------------------------------------------------------------------
// Weight pre-convert: W f32 [D][K][N]  ->  Wt f16 [D][N][K]  (transposed)
// ---------------------------------------------------------------------------
__global__ void ppnet_convert_wT(const float* __restrict__ W,
                                 _Float16* __restrict__ Wt, int K, int N)
{
    int idx = blockIdx.x * blockDim.x + threadIdx.x;     // over D*K*N
    if (idx >= D_DOM * K * N) return;
    int n = idx % N;
    int k = (idx / N) % K;
    int d = idx / (N * K);
    Wt[((size_t)d * N + n) * K + k] = (_Float16)W[idx];
}

// ---------------------------------------------------------------------------
// Batched WMMA GEMM: out[d][m][n] = act( A[d?] @ W[d] + bias[d] )
// A:  f16 row-major [.., M, K]  (a_batch_stride elements per domain; 0=shared)
// Wt: f16 [D][N][K]  (pre-transposed -> B fragments load directly from global)
// Block: 256 thr = 8 waves; tile M=128 (16/wave) x N=32 (two 16x16 per wave).
// A tile double-buffered in LDS via global_load_async_to_lds_b128.
// MODE: 0 = none, 1 = relu, 2 = 2*sigmoid
// ---------------------------------------------------------------------------
template <int MODE>
__global__ __launch_bounds__(256)
void ppnet_gemm_wmma(const _Float16* __restrict__ A, size_t a_batch_stride,
                     const _Float16* __restrict__ Wt,
                     const float* __restrict__ bias,
                     _Float16* __restrict__ out, int M, int K, int N)
{
    __shared__ _Float16 aT[2][128 * 40];   // 128 rows x 32 halves, stride 40

    const int d    = blockIdx.z;
    const int m0   = blockIdx.x * 128;
    const int n0   = blockIdx.y * 32;
    const int tid  = threadIdx.x;
    const int wave = tid >> 5;
    const int lane = tid & 31;
    const int g    = lane >> 4;    // K-half selector (ISA 16-bit A/B layouts)
    const int r    = lane & 15;    // A row / B column within tile

    const _Float16* Ad  = A  + (size_t)d * a_batch_stride;
    const _Float16* Wd  = Wt + (size_t)d * N * K;

    // staging geometry: each thread async-copies 32B (two b128) per K-step
    const int srow  = tid >> 1;
    const int skoff = (tid & 1) * 16;
    const _Float16* srcRow = Ad + (size_t)(m0 + srow) * K + skoff;

    v8f acc0 = {0.f, 0.f, 0.f, 0.f, 0.f, 0.f, 0.f, 0.f};
    v8f acc1 = {0.f, 0.f, 0.f, 0.f, 0.f, 0.f, 0.f, 0.f};

    const int nsteps = K >> 5;

    // prologue: stage tile 0
    {
        unsigned lds = (unsigned)(uintptr_t)&aT[0][srow * 40 + skoff];
        async_copy_b128(lds,      srcRow);
        async_copy_b128(lds + 16, srcRow + 8);
    }
    wait_asynccnt0();
    __syncthreads();

    for (int ks = 0; ks < nsteps; ++ks) {
        const int buf = ks & 1;

        // issue async copy of next tile into the other buffer
        if (ks + 1 < nsteps) {
            unsigned lds = (unsigned)(uintptr_t)&aT[buf ^ 1][srow * 40 + skoff];
            const _Float16* s = srcRow + (ks + 1) * 32;
            async_copy_b128(lds,      s);
            async_copy_b128(lds + 16, s + 8);
        }

        // B fragments straight from global f16 W^T (hot in L1/L2):
        // lane group g holds K = 16*g + 0..15, column n = r
        const _Float16* w0 = Wd + (size_t)(n0 + r) * K + ks * 32 + 16 * g;
        v16h b0 = load_v16h(w0);
        v16h b1 = load_v16h(w0 + (size_t)16 * K);

        // A fragment (contiguous k order, not the ISA table order):
        // halves 0..15 -> K = 16*g + 0..15 (matches B fragment order)
        const _Float16* arow = &aT[buf][(wave * 16 + r) * 40];
        v8h a_lo = *(const v8h*)&arow[16 * g];
        v8h a_hi = *(const v8h*)&arow[16 * g + 8];
        v16h a = __builtin_shufflevector(a_lo, a_hi,
                 0, 1, 2, 3, 4, 5, 6, 7, 8, 9, 10, 11, 12, 13, 14, 15);

        acc0 = __builtin_amdgcn_wmma_f32_16x16x32_f16(
                   false, a, false, b0, (short)0, acc0, false, false);
        acc1 = __builtin_amdgcn_wmma_f32_16x16x32_f16(
                   false, a, false, b1, (short)0, acc1, false, false);

        if (ks + 1 < nsteps) wait_asynccnt0();
        __syncthreads();
    }

    // epilogue: C/D layout -> lane holds column n = r, rows m = 8*g + i
    const size_t base = (size_t)d * M * (size_t)N;
#pragma unroll
    for (int t = 0; t < 2; ++t) {
        const int   n_g = n0 + 16 * t + r;
        const float bv  = bias[d * N + n_g];
        const v8f&  acc = (t == 0) ? acc0 : acc1;
#pragma unroll
        for (int i = 0; i < 8; ++i) {
            int   m = m0 + wave * 16 + g * 8 + i;
            float x = acc[i] + bv;
            if (MODE == 1)      x = fmaxf(x, 0.f);
            else if (MODE == 2) x = 2.f / (1.f + __expf(-x));
            out[base + (size_t)m * N + n_g] = (_Float16)x;
        }
    }
}

// ---------------------------------------------------------------------------
// BN statistics (sum / sumsq per (d, channel) over batch axis)
// ---------------------------------------------------------------------------
__global__ void ppnet_zero_stats(float* stats)      // 2 * D * 256 floats
{
    int i = blockIdx.x * blockDim.x + threadIdx.x;
    if (i < 2 * D_DOM * 256) stats[i] = 0.f;
}

__global__ void ppnet_bn_stats(const _Float16* __restrict__ hpre,
                               float* __restrict__ sum,
                               float* __restrict__ sumsq, int C)
{
    const int d  = blockIdx.y;
    const int c  = threadIdx.x;                 // blockDim.x == C
    const int b0 = blockIdx.x * 1024;
    const _Float16* p = hpre + ((size_t)d * B_ROWS + b0) * C + c;
    float s = 0.f, s2 = 0.f;
    for (int j = 0; j < 1024; ++j) {
        float x = (float)p[(size_t)j * C];
        s += x; s2 += x * x;
    }
    atomicAdd(&sum[d * C + c], s);
    atomicAdd(&sumsq[d * C + c], s2);
}

// ---------------------------------------------------------------------------
// BN apply: h = relu((x-mu)*rsqrt(var+eps)*gamma + beta) * gate   -> f16
// ---------------------------------------------------------------------------
__global__ void ppnet_bn_apply(const _Float16* __restrict__ hpre,
                               const _Float16* __restrict__ gate,
                               const float* __restrict__ sum,
                               const float* __restrict__ sumsq,
                               const float* __restrict__ bn_g,
                               const float* __restrict__ bn_b,
                               _Float16* __restrict__ hout, int C)
{
    size_t i = (size_t)blockIdx.x * blockDim.x + threadIdx.x;
    size_t total = (size_t)D_DOM * B_ROWS * C;
    if (i >= total) return;
    int c = (int)(i % C);
    int d = (int)(i / ((size_t)B_ROWS * C));
    const float inv_b = 1.f / (float)B_ROWS;
    float mean = sum[d * C + c] * inv_b;
    float var  = sumsq[d * C + c] * inv_b - mean * mean;
    float x = (float)hpre[i];
    float y = (x - mean) * rsqrtf(var + 1e-5f) * bn_g[d * C + c] + bn_b[d * C + c];
    y = fmaxf(y, 0.f) * (float)gate[i];
    hout[i] = (_Float16)y;
}

// ---------------------------------------------------------------------------
// Final head: only each sample's own domain matters after the select.
// ---------------------------------------------------------------------------
__global__ void ppnet_final(const _Float16* __restrict__ h,
                            const float* __restrict__ finW,
                            const float* __restrict__ finb,
                            const int* __restrict__ domain_id,
                            float* __restrict__ out)
{
    int b = blockIdx.x * blockDim.x + threadIdx.x;
    if (b >= B_ROWS) return;
    int d = domain_id[b];
    const _Float16* hp = h + ((size_t)d * B_ROWS + b) * 64;
    const float*    w  = finW + d * 64;
    float acc = finb[d];
#pragma unroll
    for (int c = 0; c < 64; ++c) acc += (float)hp[c] * w[c];
    out[b] = 1.f / (1.f + __expf(-acc));
}

// ---------------------------------------------------------------------------
extern "C" void kernel_launch(void* const* d_in, const int* in_sizes, int n_in,
                              void* d_out, int out_size, void* d_ws, size_t ws_size,
                              hipStream_t stream)
{
    (void)in_sizes; (void)n_in; (void)out_size; (void)ws_size;

    const int*   id_idx     = (const int*)d_in[0];
    const int*   agn_idx    = (const int*)d_in[1];
    const int*   domain_id  = (const int*)d_in[2];
    const float* id_tables  = (const float*)d_in[3];
    const float* agn_tables = (const float*)d_in[4];
    const float *mlp_W[3], *mlp_b[3], *bn_g[3], *bn_b[3];
    const float *gW1[3], *gb1[3], *gW2[3], *gb2[3];
    for (int i = 0; i < 3; ++i) {
        int base = 5 + i * 8;
        mlp_W[i] = (const float*)d_in[base + 0];
        mlp_b[i] = (const float*)d_in[base + 1];
        bn_g[i]  = (const float*)d_in[base + 2];
        bn_b[i]  = (const float*)d_in[base + 3];
        gW1[i]   = (const float*)d_in[base + 4];
        gb1[i]   = (const float*)d_in[base + 5];
        gW2[i]   = (const float*)d_in[base + 6];
        gb2[i]   = (const float*)d_in[base + 7];
    }
    const float* fin_W = (const float*)d_in[29];
    const float* fin_b = (const float*)d_in[30];
    float* out = (float*)d_out;

    // ---- workspace partition (256B aligned chunks) ----
    char* wsp = (char*)d_ws;
    size_t off = 0;
    auto take = [&](size_t bytes) -> char* {
        char* p = wsp + off;
        off += (bytes + 255) & ~(size_t)255;
        return p;
    };
    const size_t actsz = (size_t)D_DOM * B_ROWS * 256 * sizeof(_Float16);
    const size_t wtsz  = (size_t)D_DOM * 256 * 256 * sizeof(_Float16);
    _Float16* gate_in = (_Float16*)take((size_t)B_ROWS * 256 * sizeof(_Float16));
    _Float16* g1      = (_Float16*)take(actsz);
    _Float16* gatebuf = (_Float16*)take(actsz);
    _Float16* hpre    = (_Float16*)take(actsz);
    _Float16* hA      = (_Float16*)take(actsz);
    _Float16* hB      = (_Float16*)take(actsz);
    _Float16* wt1     = (_Float16*)take(wtsz);
    _Float16* wt2     = (_Float16*)take(wtsz);
    _Float16* wt3     = (_Float16*)take(wtsz);
    float*    stats   = (float*)take(2 * D_DOM * 256 * sizeof(float));
    float*    ssum    = stats;
    float*    ssq     = stats + D_DOM * 256;

    // ---- 1) embeddings ----
    ppnet_embed<<<(B_ROWS * NFEAT + 255) / 256, 256, 0, stream>>>(
        id_idx, agn_idx, id_tables, agn_tables, gate_in);

    // ---- 2) three gated MLP + BN layers ----
    const int FCNl[3] = {256, 128, 64};
    _Float16* hbufs[2] = {hA, hB};
    _Float16* hprev = nullptr;
    for (int i = 0; i < 3; ++i) {
        const int N    = FCNl[i];
        const int dinM = (i == 0) ? 256 : FCNl[i - 1];

        // pre-convert this layer's weights to f16 W^T (tiny)
        ppnet_convert_wT<<<(D_DOM * 256 * N + 255) / 256, 256, 0, stream>>>(
            gW1[i], wt1, 256, N);
        ppnet_convert_wT<<<(D_DOM * N * N + 255) / 256, 256, 0, stream>>>(
            gW2[i], wt2, N, N);
        ppnet_convert_wT<<<(D_DOM * dinM * N + 255) / 256, 256, 0, stream>>>(
            mlp_W[i], wt3, dinM, N);

        ppnet_zero_stats<<<(2 * D_DOM * 256 + 255) / 256, 256, 0, stream>>>(stats);

        // gate layer 1: relu(gate_in @ gW1 + gb1)   (A shared across domains)
        ppnet_gemm_wmma<1><<<dim3(B_ROWS / 128, N / 32, D_DOM), 256, 0, stream>>>(
            gate_in, 0, wt1, gb1[i], g1, B_ROWS, 256, N);

        // gate layer 2: 2*sigmoid(g1 @ gW2 + gb2)   (A batched per domain)
        ppnet_gemm_wmma<2><<<dim3(B_ROWS / 128, N / 32, D_DOM), 256, 0, stream>>>(
            g1, (size_t)B_ROWS * N, wt2, gb2[i], gatebuf, B_ROWS, N, N);

        // main MLP: X @ mlp_W + mlp_b  (X = gate_in for layer 0, else h_prev)
        ppnet_gemm_wmma<0><<<dim3(B_ROWS / 128, N / 32, D_DOM), 256, 0, stream>>>(
            (i == 0) ? gate_in : hprev,
            (i == 0) ? (size_t)0 : (size_t)B_ROWS * dinM,
            wt3, mlp_b[i], hpre, B_ROWS, dinM, N);

        // BN stats + apply (+relu, +gate multiply)
        ppnet_bn_stats<<<dim3(B_ROWS / 1024, D_DOM), N, 0, stream>>>(hpre, ssum, ssq, N);
        _Float16* hout = hbufs[i & 1];
        size_t total = (size_t)D_DOM * B_ROWS * N;
        ppnet_bn_apply<<<(unsigned)((total + 255) / 256), 256, 0, stream>>>(
            hpre, gatebuf, ssum, ssq, bn_g[i], bn_b[i], hout, N);
        hprev = hout;
    }

    // ---- 3) final sigmoid head with per-sample domain select ----
    ppnet_final<<<(B_ROWS + 255) / 256, 256, 0, stream>>>(
        hprev, fin_W, fin_b, domain_id, out);
}
